// LSResNet_36790689858047
// MI455X (gfx1250) — matrix operand order repaired and checked by reading, not compiled, not run on hardware
//
#include <hip/hip_runtime.h>
#include <hip/hip_bf16.h>

typedef __attribute__((ext_vector_type(16))) _Float16 v16h;
typedef __attribute__((ext_vector_type(8)))  _Float16 v8h;
typedef __attribute__((ext_vector_type(8)))  float    v8f;

#define BB     2
#define SS     2048
#define VV     128
#define WN     5
#define GG     80
#define GPAD   96      // K padded to 3 x 32 for WMMA
#define OUTD   80
#define NROT   16
#define EPSI   1e-5f
#define TWO_PI_F 6.2831853071795864769f

// One block = (batch b, weight w, tile of 16 s-rows). 256 threads = 8 waves.
__global__ __launch_bounds__(256)
void lsresnet_fused_kernel(const float* __restrict__ feat,   // [B,S,V,WN]
                           const float* __restrict__ rho,    // [B,S,V]
                           const float* __restrict__ theta,  // [B,S,V]
                           const float* __restrict__ mask,   // [B,S,V,1]
                           const float* __restrict__ mu_r,   // [WN,1,G]
                           const float* __restrict__ mu_t,   // [WN,1,G]
                           const float* __restrict__ sg_r,   // [WN,1,G]
                           const float* __restrict__ sg_t,   // [WN,1,G]
                           const float* __restrict__ Wc,     // [WN,G,OUT]
                           const float* __restrict__ bc,     // [WN,OUT]
                           float* __restrict__ out)          // [B,S,WN,OUT]
{
    // K dimension padded to 96 with explicit zeros -> unconditional b128 loads.
    __shared__ __align__(16) _Float16 descA[NROT][16][GPAD];  // 48 KB
    __shared__ __align__(16) _Float16 Wt[OUTD][GPAD];         // 15 KB, W transposed
    __shared__ float blds[OUTD];

    const int tid = threadIdx.x;
    const int s0  = blockIdx.x * 16;
    const int w   = blockIdx.y;
    const int b   = blockIdx.z;

    __builtin_prefetch(&Wc[w * GG * OUTD], 0, 0);   // global_prefetch_b8

    // ---- preload W transposed (f32 -> f16), zero K-pad rows; bias ----
    for (int i = tid; i < OUTD * GPAD; i += 256) {
        const int n = i / GPAD, k = i % GPAD;
        Wt[n][k] = (k < GG) ? (_Float16)Wc[w * GG * OUTD + k * OUTD + n]
                            : (_Float16)0.0f;
    }
    for (int i = tid; i < OUTD; i += 256)
        blds[i] = bc[w * OUTD + i];
    // zero the desc K-pad (columns 80..95) once; phase 1 only writes g<80
    for (int i = tid; i < NROT * 16 * (GPAD - GG); i += 256) {
        const int k = i >> 8, s = (i >> 4) & 15, gp = GG + (i & 15);
        descA[k][s][gp] = (_Float16)0.0f;
    }

    // ---- phase 1: gaussian soft-binning reduction over the V vertices ----
    // 16 s-rows * 80 gaussians = 1280 (s,g) pairs; 5 pairs per thread.
    // All 16 rotation accumulators in registers; rho-gaussian (k-invariant)
    // and the mask are hoisted out of the rotation loop.
    #pragma unroll
    for (int p = 0; p < 5; ++p) {
        const int pair = tid + p * 256;     // 0..1279
        const int s = pair / GG;
        const int g = pair % GG;

        const float mr  = mu_r[w * GG + g];
        const float mt  = mu_t[w * GG + g];
        const float srr = sg_r[w * GG + g];
        const float stt = sg_t[w * GG + g];
        const float inv_sr = 1.0f / (srr * srr + EPSI);
        const float inv_st = 1.0f / (stt * stt + EPSI);

        float num[NROT], den[NROT];
        #pragma unroll
        for (int k = 0; k < NROT; ++k) { num[k] = 0.0f; den[k] = 0.0f; }

        const size_t base = ((size_t)b * SS + (size_t)(s0 + s)) * VV;
        for (int v = 0; v < VV; ++v) {
            const float rv = rho[base + v];
            const float tv = theta[base + v];
            const float mv = mask[base + v];
            const float fv = feat[(base + v) * WN + w];
            const float dr = rv - mr;
            const float rg = __expf(-dr * dr * inv_sr) * mv;  // mask folded in
            const float fg = rg * fv;
            #pragma unroll
            for (int k = 0; k < NROT; ++k) {
                float th = tv + (float)k * (TWO_PI_F / (float)NROT);
                th = (th >= TWO_PI_F) ? (th - TWO_PI_F) : th;  // mod 2pi
                const float dt = th - mt;
                const float tg = __expf(-dt * dt * inv_st);
                num[k] = fmaf(fg, tg, num[k]);
                den[k] = fmaf(rg, tg, den[k]);
            }
        }
        #pragma unroll
        for (int k = 0; k < NROT; ++k)
            descA[k][s][g] = (_Float16)(num[k] / (den[k] + EPSI));
    }
    __syncthreads();

    // ---- phase 2: WMMA GEMM  cf = desc(16x96) @ W(96x80), max over rotations ----
    // Waves 0..4 each own one 16-wide output column tile; wave-uniform branch
    // keeps EXEC all-ones around v_wmma. All fragment loads are contiguous
    // 16-byte LDS reads (ds_load_b128) thanks to the zero-padded K dim.
    const int wave = tid >> 5;
    const int lane = tid & 31;
    if (wave < 5) {
        const int ntile = wave;
        const int row   = lane & 15;
        const int hi    = lane >> 4;

        v8f best;
        #pragma unroll
        for (int j = 0; j < 8; ++j) best[j] = -3.4e38f;

        for (int k = 0; k < NROT; ++k) {
            v8f acc;
            #pragma unroll
            for (int j = 0; j < 8; ++j) acc[j] = 0.0f;

            #pragma unroll
            for (int kk = 0; kk < GPAD; kk += 32) {
                // A fragment (16x32 f16): lane (row,hi) holds K runs
                // {kk+8hi..+7} and {kk+16+8hi..+7} -> two b128 loads
                const v8h a_lo = *(const v8h*)&descA[k][row][kk + 8 * hi];
                const v8h a_hi = *(const v8h*)&descA[k][row][kk + 16 + 8 * hi];
                const v16h a = __builtin_shufflevector(a_lo, a_hi,
                        0,1,2,3,4,5,6,7,8,9,10,11,12,13,14,15);
                // B fragment (32x16 f16): lane holds column n, K runs
                // {kk+8hi..+7} and {kk+16+8hi..+7} -> two b128 loads from transposed W
                const int n = ntile * 16 + row;
                const v8h b_lo = *(const v8h*)&Wt[n][kk + 8 * hi];
                const v8h b_hi = *(const v8h*)&Wt[n][kk + 16 + 8 * hi];
                const v16h bf = __builtin_shufflevector(b_lo, b_hi,
                        0,1,2,3,4,5,6,7,8,9,10,11,12,13,14,15);

                acc = __builtin_amdgcn_wmma_f32_16x16x32_f16(
                        /*neg_a=*/false, a, /*neg_b=*/false, bf,
                        /*c_mod=*/(short)0, acc, /*reuse_a=*/false, /*reuse_b=*/false);
            }
            #pragma unroll
            for (int j = 0; j < 8; ++j) best[j] = fmaxf(best[j], acc[j]);
        }

        // C/D layout: lane<16 -> (M=j, N=lane); lane>=16 -> (M=j+8, N=lane-16)
        const int ocol = ntile * 16 + row;
        const float bias = blds[ocol];
        #pragma unroll
        for (int j = 0; j < 8; ++j) {
            const int m = j + 8 * hi;
            const size_t s = (size_t)(s0 + m);
            out[(((size_t)b * SS + s) * WN + w) * OUTD + ocol] = best[j] + bias;
        }
    }
}

extern "C" void kernel_launch(void* const* d_in, const int* in_sizes, int n_in,
                              void* d_out, int out_size, void* d_ws, size_t ws_size,
                              hipStream_t stream) {
    const float* feat  = (const float*)d_in[0];
    const float* rho   = (const float*)d_in[1];
    const float* theta = (const float*)d_in[2];
    const float* mask  = (const float*)d_in[3];
    const float* mu_r  = (const float*)d_in[4];
    const float* mu_t  = (const float*)d_in[5];
    const float* sg_r  = (const float*)d_in[6];
    const float* sg_t  = (const float*)d_in[7];
    const float* Wc    = (const float*)d_in[8];
    const float* bc    = (const float*)d_in[9];
    float* out = (float*)d_out;

    dim3 grid(SS / 16, WN, BB);   // 128 x 5 x 2 = 1280 blocks
    lsresnet_fused_kernel<<<grid, 256, 0, stream>>>(
        feat, rho, theta, mask, mu_r, mu_t, sg_r, sg_t, Wc, bc, out);
}
